// Quantizer_78314433675272
// MI455X (gfx1250) — matrix operand order repaired and checked by reading, not compiled, not run on hardware
//
#include <hip/hip_runtime.h>
#include <hip/hip_bf16.h>

// Problem constants (match reference)
static constexpr int N_PTS  = 32768;
static constexpr int LVLS   = 4;
static constexpr int FEAT   = 512;
static constexpr int HID    = 512;
static constexpr int VQD    = 64;
static constexpr int CBSZ   = 4096;

typedef __attribute__((ext_vector_type(16))) __bf16 v16bf;
typedef __attribute__((ext_vector_type(8)))  __bf16 v8bf;
typedef __attribute__((ext_vector_type(8)))  float  v8f;
typedef __attribute__((ext_vector_type(4)))  int    v4i;

// CDNA5 async global->LDS copy path (ASYNCcnt-tracked), with sync fallback.
#if defined(__gfx1250__) && __has_builtin(__builtin_amdgcn_global_load_async_to_lds_b128)
#define HAVE_ASYNC_LDS 1
#else
#define HAVE_ASYNC_LDS 0
#endif

__device__ inline void async_copy16(const __bf16* g, __bf16* l) {
#if HAVE_ASYNC_LDS
  // Probe-confirmed parameter type: pointer to int __vector(4).
  __builtin_amdgcn_global_load_async_to_lds_b128((v4i*)(void*)g,
                                                 (v4i*)(void*)l, 0, 0);
#else
  (void)g; (void)l;
#endif
}

__device__ inline void wait_async0() {
#if defined(__gfx1250__) && __has_builtin(__builtin_amdgcn_s_wait_asynccnt)
  __builtin_amdgcn_s_wait_asynccnt(0);
#elif defined(__AMDGCN__)
  asm volatile("s_wait_asynccnt 0" ::: "memory");
#endif
}

// ---------------------------------------------------------------------------
// Fragment loaders in contiguous k order (k = 16*half + s), same order as the B fragments
// A 16x32 (row-major source): lanes 0-15 hold K=0..15 of row (lane&15);
// lanes 16-31 hold K=16..31.
__device__ inline v16bf load_a_frag(const __bf16* __restrict__ p) {
  v8bf lo = *(const v8bf*)p;
  v8bf hi = *(const v8bf*)(p + 8);
  v16bf r;
#pragma unroll
  for (int e = 0; e < 8; ++e) { r[e] = lo[e]; r[e + 8] = hi[e]; }
  return r;
}

// ---------------------------------------------------------------------------
// Tiled bf16 WMMA GEMM: C[M,Nn] = A[M,K] * Bt[Nn,K]^T + bias
// Block = 256 threads (8 waves). Tile BM=128 x BN=TN*16, BK=32.
// Each wave owns a 16-row strip and TN column tiles. B tile double-buffered
// in LDS via async global->LDS copies (ASYNCcnt) when available.
template <int TN, bool RELU, bool OUT_BF16, bool OUT_F32, bool MSE>
__global__ __launch_bounds__(256) void gemm_wmma(
    const __bf16* __restrict__ A, const __bf16* __restrict__ Bt,
    const float* __restrict__ bias, __bf16* __restrict__ outB,
    float* __restrict__ outF, int K, int Nn,
    const float* __restrict__ xref, int ldx, float* __restrict__ accum) {
  constexpr int BN = TN * 16;
  constexpr int BK = 32;
#if HAVE_ASYNC_LDS
  __shared__ __bf16 ldsB[2][BN * BK];
#else
  __shared__ __bf16 ldsB[1][BN * BK];
#endif

  const int tid  = threadIdx.x;
  const int wave = tid >> 5;
  const int lane = tid & 31;
  const int half = lane >> 4;
  const int l16  = lane & 15;
  const int rowBase = blockIdx.x * 128 + wave * 16;
  const int colBase = blockIdx.y * BN;

  v8f acc[TN];
#pragma unroll
  for (int t = 0; t < TN; ++t)
#pragma unroll
    for (int e = 0; e < 8; ++e) acc[t][e] = 0.f;

  const size_t arow = (size_t)(rowBase + l16) * K;

#if HAVE_ASYNC_LDS
  // Prologue: async-stage tile k0=0 into buffer 0.
  for (int c = tid; c < BN * 4; c += 256) {
    const int n = c >> 2, ks = (c & 3) * 8;
    async_copy16(Bt + (size_t)(colBase + n) * K + ks, &ldsB[0][n * BK + ks]);
  }
  int buf = 0;
  for (int k0 = 0; k0 < K; k0 += BK) {
    wait_async0();     // current tile landed in LDS
    __syncthreads();   // all waves past previous compute & past the wait
    if (k0 + BK < K) { // stage next tile into the other buffer
      const int nb = buf ^ 1;
      for (int c = tid; c < BN * 4; c += 256) {
        const int n = c >> 2, ks = (c & 3) * 8;
        async_copy16(Bt + (size_t)(colBase + n) * K + (k0 + BK) + ks,
                     &ldsB[nb][n * BK + ks]);
      }
    }
    v16bf af = load_a_frag(A + arow + k0 + half * 16);
#pragma unroll
    for (int t = 0; t < TN; ++t) {
      v16bf bf = *(const v16bf*)(&ldsB[buf][(t * 16 + l16) * BK + half * 16]);
      acc[t] = __builtin_amdgcn_wmma_f32_16x16x32_bf16(
          false, af, false, bf, (short)0, acc[t], false, false);
    }
    buf ^= 1;
  }
#else
  for (int k0 = 0; k0 < K; k0 += BK) {
    v16bf af = load_a_frag(A + arow + k0 + half * 16);
    __syncthreads();
    for (int c = tid; c < BN * 4; c += 256) {
      const int n = c >> 2, ks = (c & 3) * 8;
      const __bf16* bp = Bt + (size_t)(colBase + n) * K + k0 + ks;
      *(uint4*)(&ldsB[0][n * BK + ks]) = *(const uint4*)bp;
    }
    __syncthreads();
#pragma unroll
    for (int t = 0; t < TN; ++t) {
      v16bf bf = *(const v16bf*)(&ldsB[0][(t * 16 + l16) * BK + half * 16]);
      acc[t] = __builtin_amdgcn_wmma_f32_16x16x32_bf16(
          false, af, false, bf, (short)0, acc[t], false, false);
    }
  }
#endif

  // Epilogue. C/D layout: lane (half,l16) holds col = base + l16,
  // rows = rowBase + i + half*8 for VGPR i.
  float msum = 0.f;
#pragma unroll
  for (int t = 0; t < TN; ++t) {
    const int col = colBase + t * 16 + l16;
    const float b = bias[col];
#pragma unroll
    for (int i = 0; i < 8; ++i) {
      const int row = rowBase + i + half * 8;
      float v = acc[t][i] + b;
      if (RELU) v = v > 0.f ? v : 0.f;
      if (OUT_BF16) outB[(size_t)row * Nn + col] = (__bf16)v;
      if (OUT_F32)  outF[(size_t)row * Nn + col] = v;
      if (MSE) {
        float d = v - xref[(size_t)row * ldx + col];
        msum += d * d;
      }
    }
  }
  if (MSE) {
#pragma unroll
    for (int m = 1; m < 32; m <<= 1) msum += __shfl_xor(msum, m, 32);
    __shared__ float red[8];
    if (lane == 0) red[wave] = msum;
    __syncthreads();
    if (tid == 0) {
      float t = 0.f;
#pragma unroll
      for (int w = 0; w < 8; ++w) t += red[w];
      atomicAdd(accum, t);
    }
  }
}

// ---------------------------------------------------------------------------
// Distance + argmin: for each row of z (bf16 [N,64]) find argmin_j of
// ||cb_j||^2 - 2 z.cb_j over 4096 entries. One wave per 16-row tile.
__global__ __launch_bounds__(128) void dist_argmin(
    const __bf16* __restrict__ zb, const __bf16* __restrict__ cb,
    const float* __restrict__ cbn, int* __restrict__ idxOut) {
  const int tid  = threadIdx.x;
  const int wave = tid >> 5;
  const int lane = tid & 31;
  const int half = lane >> 4;
  const int l16  = lane & 15;
  const int rowBase = (blockIdx.x * 4 + wave) * 16;

  // z fragments, K = 64 -> two K=32 fragments, loaded once
  const __bf16* ap = zb + (size_t)(rowBase + l16) * VQD + half * 16;
  v8bf a0 = *(const v8bf*)ap;
  v8bf a1 = *(const v8bf*)(ap + 8);
  v8bf a2 = *(const v8bf*)(ap + 32);
  v8bf a3 = *(const v8bf*)(ap + 40);
  v16bf af0, af1;
#pragma unroll
  for (int e = 0; e < 8; ++e) {
    af0[e] = a0[e]; af0[e + 8] = a1[e];
    af1[e] = a2[e]; af1[e + 8] = a3[e];
  }

  float best[8];
  int   bidx[8];
#pragma unroll
  for (int i = 0; i < 8; ++i) { best[i] = 3.4e38f; bidx[i] = 0; }

  for (int j0 = 0; j0 < CBSZ; j0 += 16) {
    const int j = j0 + l16;
    const __bf16* bp = cb + (size_t)j * VQD + half * 16;
    v16bf bf0 = *(const v16bf*)bp;          // K 0..31
    v16bf bf1 = *(const v16bf*)(bp + 32);   // K 32..63
    v8f a;
#pragma unroll
    for (int e = 0; e < 8; ++e) a[e] = 0.f;
    a = __builtin_amdgcn_wmma_f32_16x16x32_bf16(false, af0, false, bf0,
                                                (short)0, a, false, false);
    a = __builtin_amdgcn_wmma_f32_16x16x32_bf16(false, af1, false, bf1,
                                                (short)0, a, false, false);
    const float cn = cbn[j];
#pragma unroll
    for (int i = 0; i < 8; ++i) {
      float s = cn - 2.f * a[i];
      if (s < best[i]) { best[i] = s; bidx[i] = j; }
    }
  }
  // argmin across the 16 lanes of each half (rows live per-half)
#pragma unroll
  for (int m = 1; m < 16; m <<= 1) {
#pragma unroll
    for (int i = 0; i < 8; ++i) {
      float ov = __shfl_xor(best[i], m, 32);
      int   oi = __shfl_xor(bidx[i], m, 32);
      if (ov < best[i] || (ov == best[i] && oi < bidx[i])) {
        best[i] = ov; bidx[i] = oi;
      }
    }
  }
  if (l16 == 0) {
#pragma unroll
    for (int i = 0; i < 8; ++i) idxOut[rowBase + i + half * 8] = bidx[i];
  }
}

// ---------------------------------------------------------------------------
// q = cb[idx] (bf16 copy for decoder); LAST level also accumulates commit SSE.
template <bool LAST>
__global__ __launch_bounds__(256) void gather_q(
    const int* __restrict__ idx, const float* __restrict__ cbf,
    const float* __restrict__ zf, __bf16* __restrict__ qb,
    float* __restrict__ accum) {
  const int g = blockIdx.x * 256 + threadIdx.x;
  const int i = g >> 6, k = g & 63;
  const float qv = cbf[(size_t)idx[i] * VQD + k];
  qb[g] = (__bf16)qv;
  if (LAST) {
    float d = qv - zf[g];
    float s = d * d;
#pragma unroll
    for (int m = 1; m < 32; m <<= 1) s += __shfl_xor(s, m, 32);
    __shared__ float red[8];
    if ((threadIdx.x & 31) == 0) red[threadIdx.x >> 5] = s;
    __syncthreads();
    if (threadIdx.x == 0) {
      float t = 0.f;
#pragma unroll
      for (int w = 0; w < 8; ++w) t += red[w];
      atomicAdd(accum, t);
    }
  }
}

// ---------------------------------------------------------------------------
// Weight / activation conversion helpers
__global__ void tcvt(const float* __restrict__ src, __bf16* __restrict__ dst,
                     int R, int C) {  // dst[c*R + r] = bf16(src[r*C + c])
  const int g = blockIdx.x * 256 + threadIdx.x;
  if (g >= R * C) return;
  const int r = g / C, c = g - r * C;
  dst[(size_t)c * R + r] = (__bf16)src[g];
}

__global__ __launch_bounds__(64) void cvt_cb(const float* __restrict__ src,
                                             __bf16* __restrict__ dst,
                                             float* __restrict__ cbn) {
  const int j = blockIdx.x, k = threadIdx.x;
  const float v = src[(size_t)j * VQD + k];
  dst[(size_t)j * VQD + k] = (__bf16)v;
  float s = v * v;
#pragma unroll
  for (int m = 1; m < 32; m <<= 1) s += __shfl_xor(s, m, 32);
  __shared__ float sh[2];
  if ((k & 31) == 0) sh[k >> 5] = s;
  __syncthreads();
  if (k == 0) cbn[j] = sh[0] + sh[1];
}

__global__ void cvt_x(const float* __restrict__ x, __bf16* __restrict__ dst,
                      int l) {
  const int g = blockIdx.x * 256 + threadIdx.x;
  const int i = g >> 9, k = g & 511;  // FEAT = 512
  dst[g] = (__bf16)x[(size_t)i * (LVLS * FEAT) + l * FEAT + k];
}

__global__ void zero2(float* a) {
  if (threadIdx.x < 2) a[threadIdx.x] = 0.f;
}

__global__ void finalize_k(const float* __restrict__ accum,
                           float* __restrict__ out) {
  if (threadIdx.x == 0) {
    out[0] = accum[0] / ((float)N_PTS * (float)FEAT * (float)LVLS);
    out[1] = 0.25f * accum[1] / ((float)N_PTS * (float)VQD) / (float)LVLS;
  }
}

// ---------------------------------------------------------------------------
extern "C" void kernel_launch(void* const* d_in, const int* in_sizes, int n_in,
                              void* d_out, int out_size, void* d_ws,
                              size_t ws_size, hipStream_t stream) {
  (void)in_sizes; (void)n_in; (void)out_size; (void)ws_size;
  const float* x        = (const float*)d_in[0];
  const float* enc_w1   = (const float*)d_in[1];
  const float* enc_b1   = (const float*)d_in[2];
  const float* enc_w2   = (const float*)d_in[3];
  const float* enc_b2   = (const float*)d_in[4];
  const float* dec_w1   = (const float*)d_in[5];
  const float* dec_b1   = (const float*)d_in[6];
  const float* dec_w2   = (const float*)d_in[7];
  const float* dec_b2   = (const float*)d_in[8];
  const float* codebook = (const float*)d_in[9];
  float* out = (float*)d_out;

  char* ws = (char*)d_ws;
  size_t off = 0;
  auto alloc = [&](size_t bytes) -> char* {
    char* p = ws + off;
    off = (off + bytes + 255) & ~(size_t)255;
    return p;
  };
  float*  accum = (float*)alloc(2 * sizeof(float));
  __bf16* actA  = (__bf16*)alloc((size_t)N_PTS * FEAT * 2);  // xl bf16
  __bf16* actB  = (__bf16*)alloc((size_t)N_PTS * HID * 2);   // h, then h2
  float*  z_f32 = (float*)alloc((size_t)N_PTS * VQD * 4);
  __bf16* z_bf  = (__bf16*)alloc((size_t)N_PTS * VQD * 2);
  __bf16* q_bf  = (__bf16*)alloc((size_t)N_PTS * VQD * 2);
  int*    idxb  = (int*)alloc((size_t)N_PTS * 4);
  __bf16* w1T   = (__bf16*)alloc((size_t)FEAT * HID * 2);   // [HID, FEAT]
  __bf16* w2T   = (__bf16*)alloc((size_t)HID * VQD * 2);    // [VQD, HID]
  __bf16* d1T   = (__bf16*)alloc((size_t)VQD * HID * 2);    // [HID, VQD]
  __bf16* d2T   = (__bf16*)alloc((size_t)HID * FEAT * 2);   // [FEAT, HID]
  __bf16* cb_bf = (__bf16*)alloc((size_t)CBSZ * VQD * 2);
  float*  cbn   = (float*)alloc((size_t)CBSZ * 4);

  zero2<<<1, 32, 0, stream>>>(accum);

  const dim3 blk128(128), blk256(256), blk64(64);
  for (int l = 0; l < LVLS; ++l) {
    // weight prep (tiny)
    tcvt<<<(FEAT * HID + 255) / 256, blk256, 0, stream>>>(
        enc_w1 + (size_t)l * FEAT * HID, w1T, FEAT, HID);
    tcvt<<<(HID * VQD + 255) / 256, blk256, 0, stream>>>(
        enc_w2 + (size_t)l * HID * VQD, w2T, HID, VQD);
    tcvt<<<(VQD * HID + 255) / 256, blk256, 0, stream>>>(
        dec_w1 + (size_t)l * VQD * HID, d1T, VQD, HID);
    tcvt<<<(HID * FEAT + 255) / 256, blk256, 0, stream>>>(
        dec_w2 + (size_t)l * HID * FEAT, d2T, HID, FEAT);
    cvt_cb<<<CBSZ, blk64, 0, stream>>>(codebook + (size_t)l * CBSZ * VQD,
                                       cb_bf, cbn);
    cvt_x<<<(N_PTS * FEAT) / 256, blk256, 0, stream>>>(x, actA, l);

    // h = relu(xl @ W1 + b1)  -> actB (bf16)
    gemm_wmma<8, true, true, false, false>
        <<<dim3(N_PTS / 128, HID / 128), blk256, 0, stream>>>(
            actA, w1T, enc_b1 + (size_t)l * HID, actB, nullptr, FEAT, HID,
            nullptr, 0, nullptr);
    // z = h @ W2 + b2  -> z_bf + z_f32
    gemm_wmma<4, false, true, true, false>
        <<<dim3(N_PTS / 128, VQD / 64), blk256, 0, stream>>>(
            actB, w2T, enc_b2 + (size_t)l * VQD, z_bf, z_f32, HID, VQD,
            nullptr, 0, nullptr);
    // nearest codebook entry
    dist_argmin<<<N_PTS / 64, blk128, 0, stream>>>(z_bf, cb_bf, cbn, idxb);
    // q = cb[idx]; commit SSE on last level
    if (l == LVLS - 1)
      gather_q<true><<<(N_PTS * VQD) / 256, blk256, 0, stream>>>(
          idxb, codebook + (size_t)l * CBSZ * VQD, z_f32, q_bf, accum);
    else
      gather_q<false><<<(N_PTS * VQD) / 256, blk256, 0, stream>>>(
          idxb, codebook + (size_t)l * CBSZ * VQD, z_f32, q_bf, accum);
    // h2 = relu(q @ D1 + b1)  -> actB (bf16)
    gemm_wmma<8, true, true, false, false>
        <<<dim3(N_PTS / 128, HID / 128), blk256, 0, stream>>>(
            q_bf, d1T, dec_b1 + (size_t)l * HID, actB, nullptr, VQD, HID,
            nullptr, 0, nullptr);
    // x_hat = h2 @ D2 + b2; fused SSE vs x[:, l, :]
    gemm_wmma<8, false, false, false, true>
        <<<dim3(N_PTS / 128, FEAT / 128), blk256, 0, stream>>>(
            actB, d2T, dec_b2 + (size_t)l * FEAT, nullptr, nullptr, HID, FEAT,
            x + (size_t)l * FEAT, LVLS * FEAT, accum);
  }
  finalize_k<<<1, 32, 0, stream>>>(accum, out);
}
